// MultiHeadAttentionBlock_73160472920481
// MI455X (gfx1250) — compile-verified, matrix operand order checked by reading, not run on hardware
//
#include <hip/hip_runtime.h>
#include <hip/hip_bf16.h>

typedef __bf16  bf16x16 __attribute__((ext_vector_type(16)));
typedef float   f32x8   __attribute__((ext_vector_type(8)));

#define D_MODEL 768
#define SEQ     4096
#define NHEADS  12
#define DK      64

union frag_u { bf16x16 v; unsigned u[8]; uint4 q[2]; };

__device__ __forceinline__ unsigned pack2bf(float a, float b) {
    union { unsigned u; __bf16 h[2]; } p;
    p.h[0] = (__bf16)a; p.h[1] = (__bf16)b;
    return p.u;
}

// ---------------------------------------------------------------------------
// GEMM + bias:  Y[M,N] = A[M,K] @ W[K,N] + b[N]
// Block tile 128x64, 256 threads (8 waves), wave = 32x32 (2x2 WMMA tiles),
// K-step 32. Double-buffered LDS: next slab's global loads are issued before
// the current slab's WMMAs, stored after them — one barrier per K-step and
// global latency hidden behind compute.
// ---------------------------------------------------------------------------
template <typename Tin, typename Tout>
__global__ __launch_bounds__(256) void gemm_bias_wmma(
    const Tin* __restrict__ A, const float* __restrict__ W,
    const float* __restrict__ bias, Tout* __restrict__ Y,
    int M, int N, int K)
{
    __shared__ alignas(16) __bf16 As[2][128][40];      // 80B rows
    __shared__ alignas(16) __bf16 BsP[2][16][68][2];   // [kpair][n][k&1]

    const int tid  = threadIdx.x;
    const int lane = tid & 31;
    const int wid  = tid >> 5;
    const int wm   = wid & 3;
    const int wn   = wid >> 2;
    const int h    = lane >> 4;
    const int mr   = lane & 15;

    const int m0 = blockIdx.x * 128;
    const int n0 = blockIdx.y * 64;

    float4 areg_f[4];        // f32 source staging regs
    uint4  areg_h[2];        // bf16 source staging regs
    float4 wra, wrb;

    const int akp = tid >> 4, akc = (tid & 15) * 4;    // W loader coords

    auto issue = [&](int k0) {
        if constexpr (sizeof(Tin) == 4) {
            #pragma unroll
            for (int i = 0; i < 4; ++i) {
                int idx = tid + i * 256;
                int r = idx >> 3, c = (idx & 7) * 4;
                areg_f[i] = *(const float4*)((const float*)A +
                                             (size_t)(m0 + r) * K + k0 + c);
            }
        } else {
            #pragma unroll
            for (int i = 0; i < 2; ++i) {
                int idx = tid + i * 256;
                int r = idx >> 2, c = (idx & 3) * 8;
                areg_h[i] = *(const uint4*)((const __bf16*)A +
                                            (size_t)(m0 + r) * K + k0 + c);
            }
        }
        wra = *(const float4*)(W + (size_t)(k0 + 2 * akp)     * N + n0 + akc);
        wrb = *(const float4*)(W + (size_t)(k0 + 2 * akp + 1) * N + n0 + akc);
        if (k0 + 32 < K) {   // prime L2 one more slab ahead
            __builtin_prefetch((const Tin*)A + (size_t)(m0 + (tid >> 1)) * K +
                               k0 + 32 + (tid & 1) * 16, 0, 1);
            __builtin_prefetch(W + (size_t)(k0 + 32 + (tid & 31)) * N + n0 +
                               ((tid >> 5) & 1) * 32, 0, 1);
        }
    };
    auto commit = [&](int bufw) {
        if constexpr (sizeof(Tin) == 4) {
            #pragma unroll
            for (int i = 0; i < 4; ++i) {
                int idx = tid + i * 256;
                int r = idx >> 3, c = (idx & 7) * 4;
                uint2 d; d.x = pack2bf(areg_f[i].x, areg_f[i].y);
                         d.y = pack2bf(areg_f[i].z, areg_f[i].w);
                *(uint2*)&As[bufw][r][c] = d;
            }
        } else {
            #pragma unroll
            for (int i = 0; i < 2; ++i) {
                int idx = tid + i * 256;
                int r = idx >> 2, c = (idx & 3) * 8;
                *(uint4*)&As[bufw][r][c] = areg_h[i];
            }
        }
        uint4 d;
        d.x = pack2bf(wra.x, wrb.x); d.y = pack2bf(wra.y, wrb.y);
        d.z = pack2bf(wra.z, wrb.z); d.w = pack2bf(wra.w, wrb.w);
        *(uint4*)&BsP[bufw][akp][akc][0] = d;
    };

    f32x8 acc[2][2] = {};

    issue(0); commit(0);
    __syncthreads();

    int buf = 0;
    for (int k0 = 0; k0 < K; k0 += 32, buf ^= 1) {
        const bool more = (k0 + 32 < K);
        if (more) issue(k0 + 32);

        frag_u af[2], bf[2];
        #pragma unroll
        for (int s = 0; s < 2; ++s) {
            int row = wm * 32 + s * 16 + mr;
            af[s].q[0] = *(const uint4*)&As[buf][row][h * 8];
            af[s].q[1] = *(const uint4*)&As[buf][row][16 + h * 8];
        }
        #pragma unroll
        for (int s = 0; s < 2; ++s) {
            int col = wn * 32 + s * 16 + mr;
            #pragma unroll
            for (int i = 0; i < 8; ++i) {
                int kkp = (i & 3) + 4 * h + 8 * (i >> 2);
                bf[s].u[i] = *(const unsigned*)&BsP[buf][kkp][col][0];
            }
        }
        #pragma unroll
        for (int sm = 0; sm < 2; ++sm)
            #pragma unroll
            for (int sn = 0; sn < 2; ++sn)
                acc[sm][sn] = __builtin_amdgcn_wmma_f32_16x16x32_bf16(
                    false, af[sm].v, false, bf[sn].v, (short)0, acc[sm][sn],
                    false, false);

        if (more) commit(buf ^ 1);
        __syncthreads();
    }

    #pragma unroll
    for (int sm = 0; sm < 2; ++sm)
        #pragma unroll
        for (int sn = 0; sn < 2; ++sn) {
            int col = n0 + wn * 32 + sn * 16 + mr;
            float bv = bias[col];
            #pragma unroll
            for (int j = 0; j < 8; ++j) {
                int row = m0 + wm * 32 + sm * 16 + j + 8 * h;
                Y[(size_t)row * N + col] = (Tout)(acc[sm][sn][j] + bv);
            }
        }
}

// ---------------------------------------------------------------------------
// Flash attention over one (batch, head). 128 threads = 4 waves; wave owns 16
// query rows; 32-key K/V tiles double-buffered in LDS (K row-major -> b128
// fragments; V K-pair-interleaved -> b32 fragment dwords). Softmax via C->A
// layout bounce: lane owns one row's stats, one shfl_xor(16) per max/sum,
// exp(scores) in A layout IS the P fragment. One barrier per tile.
// ---------------------------------------------------------------------------
__global__ __launch_bounds__(128) void flash_attn(
    const __bf16* __restrict__ Q, const __bf16* __restrict__ Km,
    const __bf16* __restrict__ Vm, __bf16* __restrict__ Cx)
{
    __shared__ alignas(16) __bf16 Ks[2][32][72];       // 144B rows
    __shared__ alignas(16) __bf16 VsP[2][16][68][2];   // [kpair][d][k&1]
    __shared__ float Sf[4][16][33];                    // per-wave score tile
    __shared__ float Bc[4][16];                        // per-wave broadcast

    const int tid  = threadIdx.x;
    const int lane = tid & 31;
    const int wid  = tid >> 5;
    const int h    = lane >> 4;
    const int mr   = lane & 15;

    const int head = blockIdx.y % NHEADS;
    const int b    = blockIdx.y / NHEADS;
    const int q0   = blockIdx.x * 64 + wid * 16;

    const size_t base = (size_t)b * SEQ * D_MODEL + (size_t)head * DK;

    uint4 kreg[2];
    uint2 vra[2], vrb[2];
    auto issue = [&](int kt) {
        #pragma unroll
        for (int i = 0; i < 2; ++i) {
            int idx = tid + i * 128;
            int r = idx >> 3, c = (idx & 7) * 8;
            kreg[i] = *(const uint4*)(Km + base + (size_t)(kt + r) * D_MODEL + c);
        }
        #pragma unroll
        for (int i = 0; i < 2; ++i) {
            int idx = tid + i * 128;
            int kp = idx >> 4, c = (idx & 15) * 4;
            size_t g = base + (size_t)(kt + 2 * kp) * D_MODEL + c;
            vra[i] = *(const uint2*)(Vm + g);
            vrb[i] = *(const uint2*)(Vm + g + D_MODEL);
        }
        if (kt + 32 < SEQ) {   // prime L2 one more tile ahead
            int r = tid >> 2, c = (tid & 3) * 16;
            __builtin_prefetch(Km + base + (size_t)(kt + 32 + r) * D_MODEL + c, 0, 1);
            __builtin_prefetch(Vm + base + (size_t)(kt + 32 + r) * D_MODEL + c, 0, 1);
        }
    };
    auto commit = [&](int bufw) {
        #pragma unroll
        for (int i = 0; i < 2; ++i) {
            int idx = tid + i * 128;
            int r = idx >> 3, c = (idx & 7) * 8;
            *(uint4*)&Ks[bufw][r][c] = kreg[i];
        }
        #pragma unroll
        for (int i = 0; i < 2; ++i) {
            int idx = tid + i * 128;
            int kp = idx >> 4, c = (idx & 15) * 4;
            uint4 d;
            d.x = (vra[i].x & 0xffffu) | (vrb[i].x << 16);
            d.y = (vra[i].x >> 16)     | (vrb[i].x & 0xffff0000u);
            d.z = (vra[i].y & 0xffffu) | (vrb[i].y << 16);
            d.w = (vra[i].y >> 16)     | (vrb[i].y & 0xffff0000u);
            *(uint4*)&VsP[bufw][kp][c][0] = d;
        }
    };

    // preload Q A-fragments (both 32-wide d-blocks), folding in 1/sqrt(64)
    bf16x16 qf[2];
    #pragma unroll
    for (int db = 0; db < 2; ++db)
        #pragma unroll
        for (int i = 0; i < 8; ++i) {
            int kk = (i & 3) * 2 + h * 8 + (i >> 2) * 16 + db * 32;
            const __bf16* qp = Q + base + (size_t)(q0 + mr) * D_MODEL + kk;
            qf[db][2 * i]     = (__bf16)((float)qp[0] * 0.125f);
            qf[db][2 * i + 1] = (__bf16)((float)qp[1] * 0.125f);
        }

    float m_row = -1e30f;   // stats of row (q0 + mr), replicated in both halves
    float l_row = 0.f;
    f32x8 o[4] = {};

    issue(0); commit(0);
    __syncthreads();

    int buf = 0;
    for (int kt = 0; kt < SEQ; kt += 32, buf ^= 1) {
        const bool more = (kt + 32 < SEQ);
        if (more) issue(kt + 32);

        // scores: Q(16x64) @ K^T(64x32) -> two 16x16 f32 tiles (pre-scaled)
        f32x8 sc[2] = {};
        #pragma unroll
        for (int db = 0; db < 2; ++db)
            #pragma unroll
            for (int t = 0; t < 2; ++t) {
                frag_u kf;
                kf.q[0] = *(const uint4*)&Ks[buf][t * 16 + mr][db * 32 + h * 8];
                kf.q[1] = *(const uint4*)&Ks[buf][t * 16 + mr][db * 32 + 16 + h * 8];
                sc[t] = __builtin_amdgcn_wmma_f32_16x16x32_bf16(
                    false, qf[db], false, kf.v, (short)0, sc[t], false, false);
            }

        // C-layout -> LDS (row j+8h, col mr / 16+mr)
        #pragma unroll
        for (int j = 0; j < 8; ++j) {
            int r = j + 8 * h;
            Sf[wid][r][mr]      = sc[0][j];
            Sf[wid][r][16 + mr] = sc[1][j];
        }
        // A-layout read: lane holds 16 of row mr's 32 scores
        float sv[16];
        #pragma unroll
        for (int i = 0; i < 8; ++i) {
            int kk = (i & 3) * 2 + h * 8 + (i >> 2) * 16;
            sv[2 * i]     = Sf[wid][mr][kk];
            sv[2 * i + 1] = Sf[wid][mr][kk + 1];
        }
        float mx = sv[0];
        #pragma unroll
        for (int t = 1; t < 16; ++t) mx = fmaxf(mx, sv[t]);
        mx = fmaxf(mx, __shfl_xor(mx, 16, 32));      // combine lane halves
        float mn = fmaxf(m_row, mx);
        float al = __expf(m_row - mn);
        bf16x16 pf;
        float rs = 0.f;
        #pragma unroll
        for (int t = 0; t < 16; ++t) {
            float p = __expf(sv[t] - mn);
            rs += p;
            pf[t] = (__bf16)p;                       // A-layout P fragment
        }
        rs += __shfl_xor(rs, 16, 32);
        l_row = l_row * al + rs;
        m_row = mn;

        // broadcast alpha to C-layout lanes, rescale O
        Bc[wid][mr] = al;                            // both halves write same val
        #pragma unroll
        for (int j = 0; j < 8; ++j) {
            float a = Bc[wid][j + 8 * h];
            #pragma unroll
            for (int c = 0; c < 4; ++c) o[c][j] *= a;
        }

        // O += P(16x32) @ V(32x64)
        #pragma unroll
        for (int c = 0; c < 4; ++c) {
            frag_u vf;
            #pragma unroll
            for (int i = 0; i < 8; ++i) {
                int kkp = (i & 3) + 4 * h + 8 * (i >> 2);
                vf.u[i] = *(const unsigned*)&VsP[buf][kkp][c * 16 + mr][0];
            }
            o[c] = __builtin_amdgcn_wmma_f32_16x16x32_bf16(
                false, pf, false, vf.v, (short)0, o[c], false, false);
        }

        if (more) commit(buf ^ 1);
        __syncthreads();
    }

    // broadcast row sums to C-layout lanes, normalize, store merged heads
    Bc[wid][mr] = l_row;
    #pragma unroll
    for (int j = 0; j < 8; ++j) {
        float inv = 1.f / Bc[wid][j + 8 * h];
        int s = q0 + j + 8 * h;
        #pragma unroll
        for (int c = 0; c < 4; ++c) {
            int d = c * 16 + mr;
            Cx[base + (size_t)s * D_MODEL + d] = (__bf16)(o[c][j] * inv);
        }
    }
}

extern "C" void kernel_launch(void* const* d_in, const int* in_sizes, int n_in,
                              void* d_out, int out_size, void* d_ws, size_t ws_size,
                              hipStream_t stream) {
    (void)in_sizes; (void)n_in; (void)out_size; (void)ws_size;
    const float* q   = (const float*)d_in[0];
    const float* k   = (const float*)d_in[1];
    const float* v   = (const float*)d_in[2];
    const float* w_q = (const float*)d_in[3];
    const float* b_q = (const float*)d_in[4];
    const float* w_k = (const float*)d_in[5];
    const float* b_k = (const float*)d_in[6];
    const float* w_v = (const float*)d_in[7];
    const float* b_v = (const float*)d_in[8];
    const float* w_o = (const float*)d_in[9];
    const float* b_o = (const float*)d_in[10];
    float* out = (float*)d_out;

    const int M = 2 * SEQ, N = D_MODEL, K = D_MODEL;
    const size_t mat = (size_t)M * N;              // elems per bf16 matrix
    __bf16* Qp = (__bf16*)d_ws;                    // 4 x 12.6 MB in workspace
    __bf16* Kp = Qp + mat;
    __bf16* Vp = Kp + mat;
    __bf16* Cx = Vp + mat;

    dim3 gg(M / 128, N / 64);
    gemm_bias_wmma<float, __bf16><<<gg, 256, 0, stream>>>(q, w_q, b_q, Qp, M, N, K);
    gemm_bias_wmma<float, __bf16><<<gg, 256, 0, stream>>>(k, w_k, b_k, Kp, M, N, K);
    gemm_bias_wmma<float, __bf16><<<gg, 256, 0, stream>>>(v, w_v, b_v, Vp, M, N, K);

    flash_attn<<<dim3(SEQ / 64, 2 * NHEADS), 128, 0, stream>>>(Qp, Kp, Vp, Cx);

    gemm_bias_wmma<__bf16, float><<<gg, 256, 0, stream>>>(Cx, w_o, b_o, out, M, N, K);
}
